// SparseAutoencoder_80711025426739
// MI455X (gfx1250) — compile-verified, matrix operand order checked by reading, not run on hardware
//
#include <hip/hip_runtime.h>
#include <hip/hip_bf16.h>

// ---------------------------------------------------------------------------
// Sparse Autoencoder forward for MI455X (gfx1250, wave32, WMMA + TDM).
//   pre:    x16   = f16(x - b_dec)                    (row-major, 4096x768)
//           W16sw = f16(W_enc) pre-swizzled to WMMA B-fragment order
//   z_pre   = x16 @ W16 + b_enc      WMMA f16->f32, tiles DMA'd by TDM
//   feats   = relu(topk_128(z_pre))  radix select
//   rec     = feats @ W_dec + b_dec  sparse decode (W_dec L2-resident)
// Output layout (concat, return order): [rec | feats | z_pre]
// ---------------------------------------------------------------------------

#define BATCH       4096
#define IN_DIM      768
#define FEAT_DIM    16384
#define TOPK        128

#define BM 64
#define BN 128
#define BK 32
#define KSTEPS (IN_DIM / BK)            // 24

typedef __attribute__((ext_vector_type(16))) _Float16 v16h;
typedef __attribute__((ext_vector_type(8)))  _Float16 v8h;
typedef __attribute__((ext_vector_type(8)))  float    v8f;
typedef __attribute__((ext_vector_type(4)))  unsigned v4u;
typedef __attribute__((ext_vector_type(8)))  int      v8i;
typedef __attribute__((ext_vector_type(4)))  int      v4i;

// ------------------------- TDM descriptor + issue --------------------------
// D# per CDNA5 ISA: group0 = {count, lds_addr,
// global_addr, type=2}; group1 = {data_size, tensor dims, tile dims, stride}.
// This toolchain exposes the 6-arg builtin:
//   (uint32x4 g0, int32x8 g1, int32x4 g2, int32x4 g3, int32x8 g3x, i32 cpol)
__device__ __forceinline__ void tdm_load_2d(unsigned lds_off, const void* gptr,
                                            unsigned tile_d0, unsigned tile_d1,
                                            unsigned tensor_d0, unsigned tensor_d1,
                                            unsigned stride0)
{
    unsigned long long ga = (unsigned long long)(uintptr_t)gptr;
    v4u g0;
    g0[0] = 1u;                                            // count=1, user mode
    g0[1] = lds_off;                                       // lds_addr [63:32]
    g0[2] = (unsigned)(ga & 0xFFFFFFFFu);                  // global_addr lo
    g0[3] = (unsigned)((ga >> 32) & 0x01FFFFFFu)           // global_addr hi
          | (2u << 30);                                    // type=2 ("image")
    v8i g1;
    g1[0] = (int)(1u << 16);                               // data_size=1 -> 2B
    g1[1] = (int)((tensor_d0 & 0xFFFFu) << 16);            // tensor_dim0[15:0]
    g1[2] = (int)((tensor_d0 >> 16) |
                  ((tensor_d1 & 0xFFFFu) << 16));          // td0 hi | td1 lo
    g1[3] = (int)((tensor_d1 >> 16) |
                  ((tile_d0 & 0xFFFFu) << 16));            // td1 hi | tile_dim0
    g1[4] = (int)(tile_d1 & 0xFFFFu);                      // tile_dim1 (dim2=0)
    g1[5] = (int)stride0;                                  // dim0 stride lo
    g1[6] = 0;                                             // stride hi / s1 lo
    g1[7] = 0;
    v4i z4 = {0, 0, 0, 0};
    v8i z8 = {0, 0, 0, 0, 0, 0, 0, 0};
    __builtin_amdgcn_tensor_load_to_lds(g0, g1, z4, z4, z8, 0);
}

// --------------------------- precision conversion --------------------------
__global__ __launch_bounds__(256)
void sae_cvt_x(const float* __restrict__ x, const float* __restrict__ b_dec,
               _Float16* __restrict__ x16)
{
    size_t i = (size_t)blockIdx.x * 256 + threadIdx.x;     // 0 .. 4096*768-1
    int k = (int)(i % IN_DIM);
    x16[i] = (_Float16)(x[i] - b_dec[k]);
}

// W16sw layout: for n-block nb (128 cols) and k-step kb (32 rows), an 8 KB
// contiguous block of 8 subtiles (32x16); each subtile is lane-major with the
// CDNA5 16-bit B-fragment K-swizzle:
//   lane l: n = nb*128 + st*16 + (l&15), hi = l>>4
//   half e: k = kb*32 + (e<8 ? hi*8+e : 16 + hi*8 + (e-8))
__global__ __launch_bounds__(256)
void sae_cvt_wenc(const float* __restrict__ W_enc, _Float16* __restrict__ Wsw)
{
    size_t i = (size_t)blockIdx.x * 256 + threadIdx.x;     // 0 .. 768*16384-1
    int e  = (int)(i & 15);
    int l  = (int)((i >> 4) & 31);
    int st = (int)((i >> 9) & 7);
    size_t rest = i >> 12;                                 // nb*KSTEPS + kb
    int kb = (int)(rest % KSTEPS);
    int nb = (int)(rest / KSTEPS);
    int hi = l >> 4;
    int k  = kb * 32 + ((e < 8) ? hi * 8 + e : 16 + hi * 8 + (e - 8));
    int n  = nb * 128 + st * 16 + (l & 15);
    Wsw[i] = (_Float16)W_enc[(size_t)k * FEAT_DIM + n];
}

// ------------------------------ encoder GEMM -------------------------------
// 64(M) x 128(N) block tile, K-step 32. 8 waves as 2(M) x 4(N), each wave a
// 32x32 region = 4x v_wmma_f32_16x16x32_f16. Tiles double-buffered in LDS,
// filled by the Tensor Data Mover (wave 0 issues, TENSORcnt synchronizes).
__global__ __launch_bounds__(256)
void sae_encode(const _Float16* __restrict__ x16,
                const _Float16* __restrict__ Wsw,
                const float*    __restrict__ b_enc,
                float*          __restrict__ z_pre)
{
    __shared__ _Float16 Ab[2][BM * BK];    // 2 x 4 KB, raw row-major [m][k]
    __shared__ _Float16 Bb[2][BK * BN];    // 2 x 8 KB, fragment-ordered

    const int tid  = threadIdx.x;
    const int lane = tid & 31;
    const int wave = tid >> 5;
    const int wm   = wave & 1;             // 0..1 (M)
    const int wn   = wave >> 1;            // 0..3 (N)
    const int m0   = blockIdx.y * BM;
    const int nb   = blockIdx.x;           // n-block id
    const int n0   = nb * BN;
    const int r    = lane & 15;
    const int hi   = lane >> 4;

    v8f acc[2][2] = {};

    // prologue: DMA tile 0 into buffer 0
    if (tid < 32) {
        tdm_load_2d((unsigned)(uintptr_t)&Ab[0][0],
                    x16 + (size_t)m0 * IN_DIM,
                    BK, BM, IN_DIM, BATCH, IN_DIM);
        tdm_load_2d((unsigned)(uintptr_t)&Bb[0][0],
                    Wsw + (size_t)(nb * KSTEPS) * (BK * BN),
                    BK * BN, 1, 0x40000000u, 1, BK * BN);
        __builtin_amdgcn_s_wait_tensorcnt(0);
    }
    __syncthreads();

    int cur = 0;
    for (int kb = 0; kb < KSTEPS; ++kb) {
        const int nxt = cur ^ 1;
        if (kb + 1 < KSTEPS && tid < 32) {
            tdm_load_2d((unsigned)(uintptr_t)&Ab[nxt][0],
                        x16 + (size_t)m0 * IN_DIM + (kb + 1) * BK,
                        BK, BM, IN_DIM, BATCH, IN_DIM);
            tdm_load_2d((unsigned)(uintptr_t)&Bb[nxt][0],
                        Wsw + (size_t)(nb * KSTEPS + kb + 1) * (BK * BN),
                        BK * BN, 1, 0x40000000u, 1, BK * BN);
        }

        const v8h* Av = (const v8h*)&Ab[cur][0];
        const v8h* Bv = (const v8h*)&Bb[cur][0];

        // A fragments: row mrow, chunks [K0..7 | K16..23] (hi=0) etc.
        v16h afrag[2], bfrag[2];
        #pragma unroll
        for (int mt = 0; mt < 2; ++mt) {
            int mrow = wm * 32 + mt * 16 + r;
            v8h lo = Av[mrow * 4 + hi];
            v8h hi8 = Av[mrow * 4 + 2 + hi];
            afrag[mt] = __builtin_shufflevector(lo, hi8,
                0, 1, 2, 3, 4, 5, 6, 7, 8, 9, 10, 11, 12, 13, 14, 15);
        }
        // B fragments: pre-swizzled, contiguous 2x16B per lane
        #pragma unroll
        for (int nt = 0; nt < 2; ++nt) {
            int st = wn * 2 + nt;
            v8h lo  = Bv[st * 64 + lane * 2];
            v8h hi8 = Bv[st * 64 + lane * 2 + 1];
            bfrag[nt] = __builtin_shufflevector(lo, hi8,
                0, 1, 2, 3, 4, 5, 6, 7, 8, 9, 10, 11, 12, 13, 14, 15);
        }
        #pragma unroll
        for (int mt = 0; mt < 2; ++mt)
            #pragma unroll
            for (int nt = 0; nt < 2; ++nt)
                acc[mt][nt] = __builtin_amdgcn_wmma_f32_16x16x32_f16(
                    false, afrag[mt], false, bfrag[nt],
                    (short)0, acc[mt][nt], false, false);

        if (tid < 32) __builtin_amdgcn_s_wait_tensorcnt(0);
        __syncthreads();
        cur = nxt;
    }

    // C/D layout: vgpr i -> M = i + 8*hi, lanes 0..15 -> N
    #pragma unroll
    for (int mt = 0; mt < 2; ++mt) {
        #pragma unroll
        for (int nt = 0; nt < 2; ++nt) {
            int n = n0 + wn * 32 + nt * 16 + r;
            float be = b_enc[n];
            #pragma unroll
            for (int i = 0; i < 8; ++i) {
                int m = m0 + wm * 32 + mt * 16 + i + 8 * hi;
                z_pre[(size_t)m * FEAT_DIM + n] = acc[mt][nt][i] + be;
            }
        }
    }
}

// ------------------------------ top-k select -------------------------------
__device__ __forceinline__ unsigned f2ord(float f) {
    unsigned u = __float_as_uint(f);
    return u ^ ((unsigned)((int)u >> 31) | 0x80000000u);
}

__global__ __launch_bounds__(256)
void sae_topk(const float* __restrict__ z_pre,
              float* __restrict__ features,
              int*   __restrict__ top_idx,
              float* __restrict__ top_val)
{
    __shared__ unsigned hist[256];
    __shared__ unsigned sh_prefix, sh_kneed;
    __shared__ unsigned tie_ctr, out_ctr;

    const int row = blockIdx.x;
    const int tid = threadIdx.x;
    const float* zrow = z_pre + (size_t)row * FEAT_DIM;

    if (tid < TOPK) {
        top_idx[(size_t)row * TOPK + tid] = 0;
        top_val[(size_t)row * TOPK + tid] = 0.0f;
    }
    if (tid == 0) { sh_prefix = 0u; sh_kneed = TOPK; }
    __syncthreads();

    unsigned prefix = 0u, mask = 0u, kneed = TOPK;

    #pragma unroll
    for (int shift = 24; shift >= 0; shift -= 8) {
        hist[tid] = 0u;
        __syncthreads();
        for (int i = tid; i < FEAT_DIM; i += 256) {
            unsigned u = f2ord(zrow[i]);
            if ((u & mask) == prefix)
                atomicAdd(&hist[(u >> shift) & 0xFFu], 1u);
        }
        __syncthreads();
        if (tid == 0) {
            unsigned need = sh_kneed, cum = 0u;
            int b = 255;
            for (; b >= 0; --b) { cum += hist[b]; if (cum >= need) break; }
            if (b < 0) b = 0;
            sh_prefix = sh_prefix | ((unsigned)b << shift);
            sh_kneed  = need - (cum - hist[b]);
        }
        __syncthreads();
        prefix = sh_prefix;
        kneed  = sh_kneed;
        mask  |= (0xFFu << shift);
    }

    const unsigned T = prefix;
    if (tid == 0) { tie_ctr = 0u; out_ctr = 0u; }
    __syncthreads();

    for (int i = tid; i < FEAT_DIM; i += 256) {
        float v = zrow[i];
        unsigned u = f2ord(v);
        bool sel = (u > T);
        if (!sel && u == T)
            sel = (atomicAdd(&tie_ctr, 1u) < kneed);
        float fv = sel ? fmaxf(v, 0.0f) : 0.0f;
        features[(size_t)row * FEAT_DIM + i] = fv;
        if (sel) {
            unsigned slot = atomicAdd(&out_ctr, 1u);
            top_idx[(size_t)row * TOPK + slot] = i;
            top_val[(size_t)row * TOPK + slot] = fv;
        }
    }
}

// ------------------------------ sparse decode ------------------------------
__global__ __launch_bounds__(256)
void sae_decode(const int*   __restrict__ top_idx,
                const float* __restrict__ top_val,
                const float* __restrict__ W_dec,
                const float* __restrict__ b_dec,
                float* __restrict__ rec)
{
    __shared__ int   sidx[TOPK];
    __shared__ float sval[TOPK];
    const int row = blockIdx.x;
    const int tid = threadIdx.x;

    if (tid < TOPK) {
        sidx[tid] = top_idx[(size_t)row * TOPK + tid];
        sval[tid] = top_val[(size_t)row * TOPK + tid];
    }
    __syncthreads();

    float acc0 = 0.f, acc1 = 0.f, acc2 = 0.f;
    for (int j = 0; j < TOPK; ++j) {
        float v = sval[j];
        const float* w = W_dec + (size_t)sidx[j] * IN_DIM;
        acc0 += v * w[tid];
        acc1 += v * w[tid + 256];
        acc2 += v * w[tid + 512];
    }
    size_t o = (size_t)row * IN_DIM;
    rec[o + tid]       = acc0 + b_dec[tid];
    rec[o + tid + 256] = acc1 + b_dec[tid + 256];
    rec[o + tid + 512] = acc2 + b_dec[tid + 512];
}

// -------------------------------- launcher ---------------------------------
extern "C" void kernel_launch(void* const* d_in, const int* in_sizes, int n_in,
                              void* d_out, int out_size, void* d_ws, size_t ws_size,
                              hipStream_t stream) {
    const float* x     = (const float*)d_in[0];   // (4096, 768)
    const float* W_enc = (const float*)d_in[1];   // (768, 16384)
    const float* W_dec = (const float*)d_in[2];   // (16384, 768)
    const float* b_enc = (const float*)d_in[3];   // (16384,)
    const float* b_dec = (const float*)d_in[4];   // (768,)

    float* out      = (float*)d_out;
    float* rec      = out;                                    // 4096*768
    float* features = out + (size_t)BATCH * IN_DIM;           // 4096*16384
    float* z_pre    = features + (size_t)BATCH * FEAT_DIM;    // 4096*16384

    // workspace layout (bytes):
    //   [0,   2M)  top_idx   4096*128*4
    //   [2M,  4M)  top_val   4096*128*4
    //   [4M, 10M)  x16       4096*768*2
    //   [10M,34M)  W16sw     768*16384*2
    char* ws = (char*)d_ws;
    int*      top_idx = (int*)ws;
    float*    top_val = (float*)(ws + (size_t)BATCH * TOPK * 4);
    _Float16* x16     = (_Float16*)(ws + (size_t)BATCH * TOPK * 8);
    _Float16* W16sw   = (_Float16*)(ws + (size_t)BATCH * TOPK * 8
                                       + (size_t)BATCH * IN_DIM * 2);

    // 0) precision conversion (+ encoder pre-centering, + B-fragment swizzle)
    sae_cvt_x   <<<(BATCH * IN_DIM) / 256,    256, 0, stream>>>(x, b_dec, x16);
    sae_cvt_wenc<<<(IN_DIM * FEAT_DIM) / 256, 256, 0, stream>>>(W_enc, W16sw);

    // 1) encoder GEMM (TDM-fed WMMA f16 -> f32)
    dim3 egrid(FEAT_DIM / BN, BATCH / BM);   // (128, 64)
    sae_encode<<<egrid, 256, 0, stream>>>(x16, W16sw, b_enc, z_pre);

    // 2) per-row top-128 radix select
    sae_topk<<<BATCH, 256, 0, stream>>>(z_pre, features, top_idx, top_val);

    // 3) sparse decode
    sae_decode<<<BATCH, 256, 0, stream>>>(top_idx, top_val, W_dec, b_dec, rec);
}
